// Neurocache_69913477644572
// MI455X (gfx1250) — matrix operand order repaired and checked by reading, not compiled, not run on hardware
//
#include <hip/hip_runtime.h>

// ---------------------------------------------------------------------------
// Neurocache retrieval attention for MI455X (gfx1250), wave32 + bf16 WMMA.
// H=2048, NH=32, NKV=8, HD=64, G=4, RH=1024, KVD=512, B=2, S=2048, N=16.
// Pipeline: (0) one-time fp32->bf16 conversion of reused operands,
//           (1) Q GEMM, (2) fused KV projection + attention, (3) O GEMM.
// GEMM A-tiles staged with gfx1250 async global->LDS loads (ASYNCcnt).
// ---------------------------------------------------------------------------

typedef __attribute__((ext_vector_type(16))) __bf16 v16bf;
typedef __attribute__((ext_vector_type(8)))  __bf16 v8bf;
typedef __attribute__((ext_vector_type(2)))  __bf16 v2bf;
typedef __attribute__((ext_vector_type(8)))  float  v8f;
typedef __attribute__((ext_vector_type(4)))  float  f32x4;

static constexpr int TOKENS = 4096;   // B*S
static constexpr int H      = 2048;
static constexpr int RH     = 1024;
static constexpr int KVD    = 512;
static constexpr int NNB    = 16;     // neighbors
static constexpr int NHEAD  = 32;     // query heads
static constexpr int HD     = 64;

// fp32 -> packed bf16 via native v_cvt_pk_bf16_f32.
__device__ __forceinline__ v2bf cvt2bf(float a, float b) {
#if __has_builtin(__builtin_amdgcn_cvt_pk_bf16_f32)
  return __builtin_amdgcn_cvt_pk_bf16_f32(a, b);
#else
  v2bf r; r[0] = (__bf16)a; r[1] = (__bf16)b; return r;
#endif
}
__device__ __forceinline__ __bf16 f2b(float x) { return cvt2bf(x, x)[0]; }

union BFrag { v16bf v; v8bf h[2]; v2bf p[8]; };
union BF8   { v8bf  v; v2bf p[4]; };

__device__ __forceinline__ v8bf cvt8(const float* __restrict__ src) {
  f32x4 t0 = *(const f32x4*)(src);
  f32x4 t1 = *(const f32x4*)(src + 4);
  BF8 o;
  o.p[0] = cvt2bf(t0.x, t0.y);
  o.p[1] = cvt2bf(t0.z, t0.w);
  o.p[2] = cvt2bf(t1.x, t1.y);
  o.p[3] = cvt2bf(t1.z, t1.w);
  return o.v;
}

// A fragment (16x32 bf16, M x K) from row-major bf16 LDS tile.
// Lane l: M = row_off + (l&15); halfs 0..7 -> K=klo.., halfs 8..15 -> K=klo+16..,
// klo = 8*(l>=16).  Two contiguous 16-byte runs -> ds_load_b128 x2.
__device__ __forceinline__ v16bf load_a_frag_lds(const __bf16* sm, int lda,
                                                 int row_off, int kbase, int lane) {
  int m   = row_off + (lane & 15);
  int klo = ((lane >> 4) & 1) * 8;
  const __bf16* p = sm + m * lda + kbase + klo;
  BFrag f;
  f.h[0] = *(const v8bf*)(p);
  f.h[1] = *(const v8bf*)(p + 16);
  return f.v;
}

// B fragment (32x16 bf16, K x N) from row-major bf16 weight [Nout, K].
// Lane l: N = nbase + (l&15); halfs 0..7 -> K = kbase + 8*(l>=16) + i, halfs 8..15 ->
// K = kbase + 16 + 8*(l>=16) + (i-8) (2x global_load_b128, no conversion in the hot loop).
__device__ __forceinline__ v16bf load_b_frag_bf16(const __bf16* __restrict__ W,
                                                  int ldw, int nbase, int kbase,
                                                  int lane) {
  int n   = nbase + (lane & 15);
  int khi = ((lane >> 4) & 1) * 8;
  const __bf16* p = W + (size_t)n * ldw + kbase + khi;
  BFrag f;
  f.h[0] = *(const v8bf*)(p);
  f.h[1] = *(const v8bf*)(p + 16);
  return f.v;
}

__device__ __forceinline__ v8f wmma_bf16(v16bf a, v16bf b, v8f c) {
  return __builtin_amdgcn_wmma_f32_16x16x32_bf16(false, a, false, b,
                                                 (short)0, c, false, false);
}

// ---------------------------------------------------------------------------
// Kernel 0: bulk fp32 -> bf16 conversion (vectorized, grid-stride).
// ---------------------------------------------------------------------------
__global__ __launch_bounds__(256) void cvt_f32_bf16(
    const float* __restrict__ src, __bf16* __restrict__ dst, int n8) {
  for (int g = blockIdx.x * 256 + threadIdx.x; g < n8; g += gridDim.x * 256)
    *(v8bf*)(dst + (size_t)g * 8) = cvt8(src + (size_t)g * 8);
}

// ---------------------------------------------------------------------------
// Kernels 1/3: [4096,2048] x [2048,2048]^T GEMM (K = N = 2048), all-bf16 operands.
// Block: 256 threads (8 waves). Block tile M=64, N=128; K chunked by 128 in LDS.
// Each wave: one 16-wide N-tile, 4 M-subtile accumulators (B frag reused 4x).
// A chunk staged via async global->LDS DMA (no VGPR bounce, ASYNCcnt tracked).
// ---------------------------------------------------------------------------
template <bool O_BF16>
__global__ __launch_bounds__(256) void gemm2048_wmma(
    const __bf16* __restrict__ A,    // [4096, 2048] bf16
    const __bf16* __restrict__ W,    // [2048, 2048] bf16, row-major [Nout, K]
    void* __restrict__ Optr,         // [4096, 2048]
    float scale) {
  constexpr int K = 2048, BM = 64, CK = 128;
  __shared__ __bf16 sA[BM * CK];     // 16 KB

  const int m_base = blockIdx.x * BM;
  const int tid    = threadIdx.x;
  const int wave   = tid >> 5;
  const int lane   = tid & 31;
  const int n_base = blockIdx.y * 128 + wave * 16;

  v8f acc[4] = {};

  for (int kc = 0; kc < K; kc += CK) {
    __syncthreads();
    // ---- stage A chunk (64 rows x 128 cols) via async global->LDS b128 ----
    {
      const __bf16* Ac = A + (size_t)m_base * K + kc;
#pragma unroll
      for (int it = 0; it < (BM * CK / 8) / 256; ++it) {
        int g   = it * 256 + tid;
        int row = g >> 4, c8 = (g & 15) * 8;
        unsigned     lds_off = (unsigned)(uintptr_t)(sA + row * CK + c8);
        const __bf16* gp     = Ac + (size_t)row * K + c8;
        asm volatile("global_load_async_to_lds_b128 %0, %1, off"
                     :: "v"(lds_off), "v"(gp) : "memory");
      }
      asm volatile("s_wait_asynccnt 0x0" ::: "memory");
    }
    __syncthreads();

    // prefetch next chunk of the streamed weight rows (global_prefetch_b8)
    if (kc + CK < K)
      __builtin_prefetch(W + (size_t)(n_base + (lane & 15)) * K + kc + CK +
                             ((lane >> 4) & 1) * 16, 0, 3);

    // ---- compute: 4 k-steps x 4 M-subtiles ----
#pragma unroll
    for (int ks = 0; ks < CK / 32; ++ks) {
      v16bf b = load_b_frag_bf16(W, K, n_base, kc + ks * 32, lane);
#pragma unroll
      for (int ms = 0; ms < 4; ++ms) {
        v16bf a = load_a_frag_lds(sA, CK, ms * 16, ks * 32, lane);
        acc[ms] = wmma_bf16(a, b, acc[ms]);
      }
    }
  }

  const int row0 = ((lane >> 4) & 1) * 8;
  const int n    = n_base + (lane & 15);
#pragma unroll
  for (int ms = 0; ms < 4; ++ms) {
#pragma unroll
    for (int r = 0; r < 8; ++r) {
      size_t idx = (size_t)(m_base + ms * 16 + row0 + r) * H + n;
      float val  = acc[ms][r] * scale;
      if (O_BF16) ((__bf16*)Optr)[idx] = f2b(val);
      else        ((float*)Optr)[idx]  = val;
    }
  }
}

// ---------------------------------------------------------------------------
// Kernel 2: fused K/V projection + grouped-query attention, one WG per token.
// The 16 neighbors are exactly one WMMA M-tile; its A fragment is shared by
// all 8 accumulators of a wave (8 WMMAs per A load).
// ---------------------------------------------------------------------------
__global__ __launch_bounds__(256) void kv_attn_wmma(
    const float*  __restrict__ retr,  // [4096, 16, 1024] fp32 (single-use: cvt on stage)
    const __bf16* __restrict__ Wk,    // [512, 1024] bf16
    const __bf16* __restrict__ Wv,    // [512, 1024] bf16
    const __bf16* __restrict__ q,     // [4096, 2048] bf16 (pre-scaled)
    __bf16* __restrict__ attn_out)    // [4096, 2048] bf16
{
  __shared__ __bf16 sA[NNB * RH];        // 32 KB: retrieved bf16 (reused for q/scores)
  __shared__ __bf16 sKV[2 * NNB * KVD];  // 32 KB: k then v, bf16

  const int tok  = blockIdx.x;
  const int tid  = threadIdx.x;
  const int wave = tid >> 5;
  const int lane = tid & 31;

  // ---- stage retrieved[tok] (16 x 1024) as bf16 (converting pass) ----
  {
    const float* rb = retr + (size_t)tok * NNB * RH;
    for (int g = tid; g < NNB * RH / 8; g += 256)
      *(v8bf*)(sA + g * 8) = cvt8(rb + g * 8);
  }
  __syncthreads();

  // ---- K/V projection: waves 0-3 -> K, waves 4-7 -> V; 8 N-tiles per wave ----
  const __bf16* Wt = (wave < 4) ? Wk : Wv;
  const int tbase  = (wave & 3) * 8;   // first of 8 N-tiles (each 16 channels)

  v8f acc[8] = {};
  for (int c = 0; c < RH; c += 32) {
    v16bf a = load_a_frag_lds(sA, RH, 0, c, lane);
    __builtin_prefetch(Wt + (size_t)(tbase * 16 + (lane & 15)) * RH + c + 32, 0, 3);
#pragma unroll
    for (int j = 0; j < 8; ++j) {
      v16bf b = load_b_frag_bf16(Wt, RH, (tbase + j) * 16, c, lane);
      acc[j] = wmma_bf16(a, b, acc[j]);
    }
  }

  const int mat  = wave >> 2;
  const int row0 = ((lane >> 4) & 1) * 8;
#pragma unroll
  for (int j = 0; j < 8; ++j) {
    int ch = (tbase + j) * 16 + (lane & 15);
#pragma unroll
    for (int r = 0; r < 8; ++r)
      sKV[mat * (NNB * KVD) + (row0 + r) * KVD + ch] = f2b(acc[j][r]);
  }
  __syncthreads();

  // ---- retrieved tile is dead: alias LDS for q (2048 f32) + scores (32x16) ----
  float* qf = (float*)sA;
  float* sc = qf + H;
  {
    const __bf16* qg = q + (size_t)tok * H;
    for (int g = tid; g < H / 8; g += 256) {
      v8bf t = *(const v8bf*)(qg + g * 8);
#pragma unroll
      for (int e = 0; e < 8; ++e) qf[g * 8 + e] = (float)t[e];
    }
  }
  __syncthreads();

  // ---- scores: 512 (head, neighbor) pairs -> 2 per thread ----
#pragma unroll
  for (int j = 0; j < 2; ++j) {
    const int p  = tid * 2 + j;
    const int h  = p >> 4;
    const int n  = p & 15;
    const int kv = h >> 2;
    const __bf16* kp = sKV + n * KVD + kv * HD;
    const float*  qp = qf + h * HD;
    float d = 0.f;
#pragma unroll 8
    for (int e = 0; e < HD; ++e) d += qp[e] * (float)kp[e];
    sc[h * NNB + n] = d;
  }
  __syncthreads();

  // ---- softmax over the 16 neighbors, one head per thread ----
  if (tid < NHEAD) {
    const int h = tid;
    float mx = -1e30f;
#pragma unroll
    for (int n = 0; n < NNB; ++n) mx = fmaxf(mx, sc[h * NNB + n]);
    float ex[NNB], s = 0.f;
#pragma unroll
    for (int n = 0; n < NNB; ++n) { ex[n] = __expf(sc[h * NNB + n] - mx); s += ex[n]; }
    const float inv = 1.f / s;
#pragma unroll
    for (int n = 0; n < NNB; ++n) sc[h * NNB + n] = ex[n] * inv;
  }
  __syncthreads();

  // ---- weighted sum of V: head h = tid/8, dims d0..d0+7 ----
  const int h  = tid >> 3;
  const int d0 = (tid & 7) * 8;
  const int kv = h >> 2;
  __bf16* og = attn_out + (size_t)tok * H + h * HD + d0;
#pragma unroll
  for (int dd = 0; dd < 8; ++dd) {
    float a = 0.f;
#pragma unroll
    for (int n = 0; n < NNB; ++n)
      a += sc[h * NNB + n] * (float)sKV[NNB * KVD + n * KVD + kv * HD + d0 + dd];
    og[dd] = f2b(a);
  }
}

// ---------------------------------------------------------------------------
extern "C" void kernel_launch(void* const* d_in, const int* in_sizes, int n_in,
                              void* d_out, int out_size, void* d_ws, size_t ws_size,
                              hipStream_t stream) {
  (void)in_sizes; (void)n_in; (void)out_size; (void)ws_size;

  const float* hs   = (const float*)d_in[0];   // [2,2048,2048]
  const float* retr = (const float*)d_in[1];   // [2,2048,16,1024]
  const float* Wq   = (const float*)d_in[2];   // [2048,2048]
  const float* Wk   = (const float*)d_in[3];   // [512,1024]
  const float* Wv   = (const float*)d_in[4];   // [512,1024]
  const float* Wo   = (const float*)d_in[5];   // [2048,2048]
  float* out        = (float*)d_out;           // [2,2048,2048]

  // workspace layout (bf16 elements)
  __bf16* q_ws  = (__bf16*)d_ws;                        // TOKENS*H
  __bf16* ao_ws = q_ws  + (size_t)TOKENS * H;           // TOKENS*H
  __bf16* hs_bf = ao_ws + (size_t)TOKENS * H;           // TOKENS*H
  __bf16* Wq_bf = hs_bf + (size_t)TOKENS * H;           // H*H
  __bf16* Wo_bf = Wq_bf + (size_t)H * H;                // H*H
  __bf16* Wk_bf = Wo_bf + (size_t)H * H;                // KVD*RH
  __bf16* Wv_bf = Wk_bf + (size_t)KVD * RH;             // KVD*RH

  const dim3 blk(256);

  // 0) one-time fp32 -> bf16 conversion of all reused operands
  cvt_f32_bf16<<<dim3(2048), blk, 0, stream>>>(hs, hs_bf, TOKENS * H / 8);
  cvt_f32_bf16<<<dim3(1024), blk, 0, stream>>>(Wq, Wq_bf, H * H / 8);
  cvt_f32_bf16<<<dim3(256),  blk, 0, stream>>>(Wk, Wk_bf, KVD * RH / 8);
  cvt_f32_bf16<<<dim3(256),  blk, 0, stream>>>(Wv, Wv_bf, KVD * RH / 8);
  cvt_f32_bf16<<<dim3(1024), blk, 0, stream>>>(Wo, Wo_bf, H * H / 8);

  const dim3 gemm_grid(TOKENS / 64, H / 128);           // (64, 16)

  // 1) Q = hs . Wq^T * HD^-0.5   (scale folded in; q stored bf16)
  gemm2048_wmma<true><<<gemm_grid, blk, 0, stream>>>(hs_bf, Wq_bf, q_ws, 0.125f);

  // 2) fused K/V projection + attention per token
  kv_attn_wmma<<<dim3(TOKENS), blk, 0, stream>>>(retr, Wk_bf, Wv_bf, q_ws, ao_ws);

  // 3) out = attn . Wo^T
  gemm2048_wmma<false><<<gemm_grid, blk, 0, stream>>>(ao_ws, Wo_bf, out, 1.0f);
}
